// ACTTransformerLayer_29712583754347
// MI455X (gfx1250) — matrix operand order repaired and checked by reading, not compiled, not run on hardware
//
#include <hip/hip_runtime.h>

typedef __attribute__((ext_vector_type(16))) __bf16 v16bf;
typedef __attribute__((ext_vector_type(8)))  __bf16 v8bf;
typedef __attribute__((ext_vector_type(8)))  float  v8f;

#define D_DIM   1024
#define M_DIM   16384   // B*T = 8*2048
#define L_STEPS 4
#define THRESH  0.99f
#define KTILES  32      // 1024 / 32
#define BTILE_ELEMS 4096  // 128 cols * 32 k of bf16 per buffer

__device__ __forceinline__ float gelu_tanh(float x) {
    float x3 = x * x * x;
    return 0.5f * x * (1.0f + tanhf(0.7978845608028654f * (x + 0.044715f * x3)));
}

// CDNA5 split workgroup barrier WITHOUT the compiler's loadcnt-0 drain:
// my async->LDS writes must be complete (asynccnt), then signal/wait.
// ds reads of the current buffer were already consumed by the WMMAs issued
// above (hardware waits DScnt before WMMA operand use), so in-flight A
// prefetch global loads may legally remain outstanding across the barrier.
__device__ __forceinline__ void stage_barrier() {
    asm volatile("s_wait_asynccnt 0x0\n\t"
                 "s_barrier_signal -1\n\t"
                 "s_barrier_wait -1" ::: "memory");
}

// ---- f32 -> bf16 cast of x into h buffer 0 -------------------------------
__global__ void cast_bf16_kernel(const float* __restrict__ src,
                                 __bf16* __restrict__ dst, int n) {
    int i = blockIdx.x * blockDim.x + threadIdx.x;
    if (i < n) dst[i] = (__bf16)src[i];
}

// ---- swizzle layer_w (L,D,D) f32 into WMMA B-fragment-linear bf16 --------
// chunk index = ((step*64 + ntile)*32 + ktile)*32 + lane, 16 bf16 per chunk.
// lane<16:  col n = ntile*16+lane,      K = ktile*32 + 0..15
// lane>=16: col n = ntile*16+(lane-16), K = ktile*32 + 16..31
__global__ void swizzle_w_kernel(const float* __restrict__ W,
                                 __bf16* __restrict__ Bsw) {
    int idx  = blockIdx.x * blockDim.x + threadIdx.x;   // 262144 total
    int lane = idx & 31;
    int jn   = (idx >> 10) & 63;
    int s    = idx >> 16;
    const float* Ws = W + (size_t)s * D_DIM * D_DIM;
    int n     = jn * 16 + (lane & 15);
    int kbase = ((idx >> 5) & 31) * 32 + ((lane & 16) ? 16 : 0);
    __bf16* dst = Bsw + (size_t)idx * 16;
#pragma unroll
    for (int i = 0; i < 16; ++i)
        dst[i] = (__bf16)Ws[(size_t)(kbase + i) * D_DIM + n];
}

// async-copy one 32B fragment chunk (per thread) from global to LDS
__device__ __forceinline__ void async_copy32(const __bf16* gsrc, __bf16* ldst) {
    unsigned lo = (unsigned)(size_t)ldst;   // LDS aperture 4GB aligned: low 32 bits = LDS offset
    asm volatile("global_load_async_to_lds_b128 %0, %1, off"
                 :: "v"(lo), "v"(gsrc) : "memory");
    asm volatile("global_load_async_to_lds_b128 %0, %1, off offset:16"
                 :: "v"(lo), "v"(gsrc) : "memory");
}

__device__ __forceinline__ v16bf load_a_frag(const __bf16* row, int off0, int off1) {
    v8bf lo = *(const v8bf*)(row + off0);
    v8bf hi = *(const v8bf*)(row + off1);
    v16bf a;
#pragma unroll
    for (int i = 0; i < 8; ++i) { a[i] = lo[i]; a[8 + i] = hi[i]; }
    return a;
}

// ---- fused GEMM + bias + GELU + halting partial dot ----------------------
// grid = (M/128, D/128); block = 256 = 8 wave32 arranged 4(M) x 2(N).
// Wave tile: 32 rows x 64 cols  (2 A-frags x 4 B-frags, B reused 2x from LDS)
__global__ __launch_bounds__(256)
void gemm_gelu_kernel(const __bf16* __restrict__ A,      // (M, D) bf16 row-major
                      const __bf16* __restrict__ Bsw,    // swizzled W for this step
                      const float*  __restrict__ bias,   // (D,)
                      const float*  __restrict__ hw,     // halt_w (D,)
                      __bf16* __restrict__ Hout,         // (M, D) bf16 row-major
                      float* __restrict__ halt_part) {   // (M, 16)
    __shared__ __align__(32) __bf16 smem[2 * BTILE_ELEMS];   // 16 KB, double buffered

    const int tid  = threadIdx.x;
    const int lane = tid & 31;
    const int wv   = tid >> 5;          // 0..7
    const int mi   = wv >> 1;           // 0..3
    const int ni   = wv & 1;            // 0..1
    const int lm   = lane & 15;
    const int half = lane >> 4;

    const int m0   = blockIdx.x * 128 + mi * 32;      // wave's first row
    const int jn0  = blockIdx.y * 8;                  // block's first ntile
    const int n0   = blockIdx.y * 128 + ni * 64;      // wave's first col

    // per-thread global source for cooperative B staging: thread t owns
    // ntile (t>>5), lane (t&31) of the block's 8-ntile slab.
    const __bf16* bsrc_base = Bsw +
        (((size_t)(jn0 + (tid >> 5)) * KTILES) * 32 + (tid & 31)) * 16;
    __bf16* bdst0 = smem + (size_t)tid * 16;                // buffer 0 slot
    __bf16* bdst1 = smem + BTILE_ELEMS + (size_t)tid * 16;  // buffer 1 slot

    v8f c0[4] = {};
    v8f c1[4] = {};

    const __bf16* Arow0 = A + (size_t)(m0 + lm) * D_DIM;
    const __bf16* Arow1 = A + (size_t)(m0 + 16 + lm) * D_DIM;
    const int ka0 = half ? 16 : 0;
    const int ka1 = half ? 24 : 8;

    // prologue: stage B(kt=0) into buffer 0; prefetch A(kt=0) into regs
    async_copy32(bsrc_base, bdst0);
    v16bf a0 = load_a_frag(Arow0, ka0, ka1);
    v16bf a1 = load_a_frag(Arow1, ka0, ka1);
    stage_barrier();

    // Keep this loop rolled: compiler unrolling by 2 splits the accumulator
    // register webs and inserts ~64 v_dual_mov copies + WMMA hazard NOPs per
    // K-slice. Double buffering is already done manually in source.
#pragma unroll 1
    for (int kt = 0; kt < KTILES; ++kt) {
        const int cur = kt & 1;
        // prefetch next K-slice: B -> LDS (async), A -> regs
        if (kt + 1 < KTILES) {
            async_copy32(bsrc_base + (size_t)(kt + 1) * 32 * 16,
                         cur ? bdst0 : bdst1);
        }
        const int kn = (kt + 1 < KTILES) ? (kt + 1) * 32 : kt * 32;
        v16bf na0 = load_a_frag(Arow0, kn + ka0, kn + ka1);
        v16bf na1 = load_a_frag(Arow1, kn + ka0, kn + ka1);

        const __bf16* bbuf = smem + cur * BTILE_ELEMS;
#pragma unroll
        for (int j = 0; j < 4; ++j) {
            v16bf b = *(const v16bf*)(bbuf + ((size_t)(ni * 4 + j) * 32 + lane) * 16);
            c0[j] = __builtin_amdgcn_wmma_f32_16x16x32_bf16(
                false, a0, false, b, (short)0, c0[j], false, false);
            c1[j] = __builtin_amdgcn_wmma_f32_16x16x32_bf16(
                false, a1, false, b, (short)0, c1[j], false, false);
        }
        a0 = na0; a1 = na1;
        stage_barrier();
    }

    // epilogue: bias + gelu, store bf16 h', accumulate halting partials
    float hs0[8] = {};
    float hs1[8] = {};
    const int rb0 = m0 + (half ? 8 : 0);
    const int rb1 = rb0 + 16;
#pragma unroll
    for (int j = 0; j < 4; ++j) {
        int n     = n0 + j * 16 + lm;
        float bn  = bias[n];
        float hwn = hw[n];
#pragma unroll
        for (int v = 0; v < 8; ++v) {
            float g0 = gelu_tanh(c0[j][v] + bn);
            float g1 = gelu_tanh(c1[j][v] + bn);
            Hout[(size_t)(rb0 + v) * D_DIM + n] = (__bf16)g0;
            Hout[(size_t)(rb1 + v) * D_DIM + n] = (__bf16)g1;
            hs0[v] += g0 * hwn;
            hs1[v] += g1 * hwn;
        }
    }
    const int qidx = blockIdx.y * 2 + ni;
#pragma unroll
    for (int v = 0; v < 8; ++v) {
        float s0 = hs0[v], s1 = hs1[v];
        s0 += __shfl_xor(s0, 1, 32); s1 += __shfl_xor(s1, 1, 32);
        s0 += __shfl_xor(s0, 2, 32); s1 += __shfl_xor(s1, 2, 32);
        s0 += __shfl_xor(s0, 4, 32); s1 += __shfl_xor(s1, 4, 32);
        s0 += __shfl_xor(s0, 8, 32); s1 += __shfl_xor(s1, 8, 32);
        if (lm == 0) {
            halt_part[(size_t)(rb0 + v) * 16 + qidx] = s0;
            halt_part[(size_t)(rb1 + v) * 16 + qidx] = s1;
        }
    }
}

// ---- ACT state update + weighted output accumulation ---------------------
// grid = M/8 blocks of 256; one wave32 per row.
__global__ __launch_bounds__(256)
void act_update_kernel(const __bf16* __restrict__ H,
                       const float* __restrict__ halt_part,
                       float* __restrict__ cum_ws, float* __restrict__ rem_ws,
                       float* __restrict__ pon_ws,
                       const float* __restrict__ halt_b,
                       float* __restrict__ out, int step, int is_last) {
    const int row  = blockIdx.x * 8 + (threadIdx.x >> 5);
    const int lane = threadIdx.x & 31;
    float cum = (step == 0) ? 0.f : cum_ws[row];
    float rem = (step == 0) ? 1.f : rem_ws[row];
    float acc = 0.f;
#pragma unroll
    for (int q = 0; q < 16; ++q) acc += halt_part[(size_t)row * 16 + q];
    float p = 1.f / (1.f + expf(-(acc + halt_b[0])));
    float weight = is_last ? rem : (((cum + p) >= THRESH) ? rem : p);
    float ncum = cum + weight;               // contrib == weight in all branches
    float nrem = fmaxf(0.f, 1.f - ncum);
    if (lane == 0) {
        cum_ws[row] = ncum;
        rem_ws[row] = nrem;
        pon_ws[row] = ((step == 0) ? 0.f : pon_ws[row]) + weight;
    }
    size_t base = (size_t)row * D_DIM;
    for (int d = lane; d < D_DIM; d += 32) {
        float h = (float)H[base + d];
        float o = (step == 0) ? 0.f : out[base + d];
        out[base + d] = o + weight * h;
    }
}

// ---- ponder_cost = mean(ponder_steps) ------------------------------------
__global__ void ponder_kernel(const float* __restrict__ pon_ws,
                              float* __restrict__ out) {
    __shared__ float sm[256];
    float s = 0.f;
    for (int i = threadIdx.x; i < M_DIM; i += 256) s += pon_ws[i];
    sm[threadIdx.x] = s;
    __syncthreads();
    for (int k = 128; k > 0; k >>= 1) {
        if (threadIdx.x < k) sm[threadIdx.x] += sm[threadIdx.x + k];
        __syncthreads();
    }
    if (threadIdx.x == 0) out[0] = sm[0] / (float)M_DIM;
}

extern "C" void kernel_launch(void* const* d_in, const int* in_sizes, int n_in,
                              void* d_out, int out_size, void* d_ws, size_t ws_size,
                              hipStream_t stream) {
    const float* x       = (const float*)d_in[0];   // (8,2048,1024)
    const float* layer_w = (const float*)d_in[1];   // (4,1024,1024)
    const float* layer_b = (const float*)d_in[2];   // (4,1024)
    const float* halt_w  = (const float*)d_in[3];   // (1024,1)
    const float* halt_b  = (const float*)d_in[4];   // (1,)
    float* out = (float*)d_out;                     // 16777216 + 1 floats

    char* ws = (char*)d_ws;
    __bf16* Bsw      = (__bf16*)(ws);                          //  8 MiB
    __bf16* h0       = (__bf16*)(ws + ((size_t)8  << 20));     // 32 MiB
    __bf16* h1       = (__bf16*)(ws + ((size_t)40 << 20));     // 32 MiB
    float*  halt_prt = (float*)(ws + ((size_t)72 << 20));      // 1 MiB
    float*  cum_ws   = (float*)(ws + ((size_t)73 << 20));
    float*  rem_ws   = cum_ws + M_DIM;
    float*  pon_ws   = rem_ws + M_DIM;

    // 1) cast x to bf16
    cast_bf16_kernel<<<(M_DIM * D_DIM) / 256, 256, 0, stream>>>(
        x, h0, M_DIM * D_DIM);
    // 2) swizzle all 4 weight matrices into WMMA B-fragment layout
    swizzle_w_kernel<<<1024, 256, 0, stream>>>(layer_w, Bsw);

    __bf16* bufs[2] = {h0, h1};
    for (int s = 0; s < L_STEPS; ++s) {
        gemm_gelu_kernel<<<dim3(M_DIM / 128, D_DIM / 128), 256, 0, stream>>>(
            bufs[s & 1],
            Bsw + (size_t)s * D_DIM * D_DIM,
            layer_b + (size_t)s * D_DIM,
            halt_w,
            bufs[(s + 1) & 1],
            halt_prt);
        act_update_kernel<<<M_DIM / 8, 256, 0, stream>>>(
            bufs[(s + 1) & 1], halt_prt, cum_ws, rem_ws, pon_ws,
            halt_b, out, s, (s == L_STEPS - 1) ? 1 : 0);
    }
    ponder_kernel<<<1, 256, 0, stream>>>(pon_ws, out + (size_t)M_DIM * D_DIM);
}
